// Attention_45930380264337
// MI455X (gfx1250) — matrix operand order repaired and checked by reading, not compiled, not run on hardware
//
#include <hip/hip_runtime.h>
#include <hip/hip_bf16.h>

// ---------------------------------------------------------------------------
// Causal attention for B=4, S=4096, E=1024, D=128 (fp32 in/out).
// Phase 0: pre-pack WQ/WK/WV into f16 WMMA B-fragment layout (once, 768 KB),
//          with all 8 fragments of a (chunk,lane) contiguous (256 B/lane).
// Phase 1: WMMA f16 GEMM producing Qh (pre-scaled by 1/sqrt(E)), Kh, VhT.
//          All 8 B-fragments preloaded per k-chunk; sched_barrier pins the
//          load clause ahead of the 8 WMMAs -> progressive waits, not 8 RTs.
// Phase 2: flash-attention, one wave per 16-query tile:
//          S^T = K @ Q^T  (A=K rows, B=Q^T, contiguous [S,D] f16 loads)
//          online softmax per query (query == lane&15 in the S^T C-layout)
//          P^T -> LDS (f16) -> reread as A-fragment; O += P @ V with V^T
//          stored [D,S].  K and V fragments for the whole 32-key block are
//          preloaded into registers up front (V loads overlap GEMM1+softmax),
//          and the *next* K/V blocks are prefetched via the CDNA5 Tensor Data
//          Mover (tensor_load_to_lds + s_wait_tensorcnt).
// Both WMMA kernels use __launch_bounds__(threads, 1): at ~1 wave/SIMD on
// MI455X, occupancy is irrelevant and the full VGPR file should be spent on
// keeping fragments resident.
// ---------------------------------------------------------------------------

typedef __attribute__((ext_vector_type(16))) _Float16 v16h;
typedef __attribute__((ext_vector_type(8)))  _Float16 h8;
typedef __attribute__((ext_vector_type(8)))  float    v8f;
typedef unsigned int u32;
typedef u32 u32x4 __attribute__((ext_vector_type(4)));
typedef int i32x4 __attribute__((ext_vector_type(4)));
typedef int i32x8 __attribute__((ext_vector_type(8)));

constexpr int BB = 4;
constexpr int SS = 4096;
constexpr int EE = 1024;
constexpr int DD = 128;
constexpr float SCALE = 0.03125f;  // 1/sqrt(1024)

constexpr int ECHUNKS = EE / 32;                       // 32 k-chunks of 32
constexpr int WP_PER_W = ECHUNKS * 32 * 8 * 16;        // halfs per weight matrix
// Wpack[which][ec][lane][dt][i] , i=0..15 : half i of lane's B-fragment dt
// -> per (ec,lane): 8 fragments x 32B contiguous = 256 B

static __device__ __forceinline__ v8f wmma_f16(const v16h& a, const v16h& b, const v8f& c) {
  return __builtin_amdgcn_wmma_f32_16x16x32_f16(false, a, false, b, (short)0, c, false, false);
}

// Forbid the scheduler from sinking preloaded fragments past this point.
static __device__ __forceinline__ void sched_fence() {
#if __has_builtin(__builtin_amdgcn_sched_barrier)
  __builtin_amdgcn_sched_barrier(0);
#endif
}

// ---------------------------------------------------------------------------
// TDM 2D tile prefetch: global (row_stride halfs between rows) -> LDS.
// D# packing per CDNA5 ISA 8.3/8.4: group0 = {count=1 | lds_addr | global_addr
// | type=2}, group1 = {data_size=2B, tensor_dim0=row_stride, tensor_dim1=
// tile_h, tile_dim0=tile_w, tile_dim1=tile_h, dim0_stride=row_stride}.
// ---------------------------------------------------------------------------
static __device__ __forceinline__ void tdm_load_2d(const void* gaddr, u32 lds_off,
                                                   u32 tile_w, u32 tile_h,
                                                   u32 row_stride) {
#if __has_builtin(__builtin_amdgcn_tensor_load_to_lds)
  const unsigned long long ga = (unsigned long long)(size_t)gaddr;
  const u32 ga_lo = (u32)__builtin_amdgcn_readfirstlane((int)(u32)ga);
  const u32 ga_hi = (u32)__builtin_amdgcn_readfirstlane((int)(u32)(ga >> 32));
  const u32 loff  = (u32)__builtin_amdgcn_readfirstlane((int)lds_off);
  u32x4 g0;
  g0[0] = 1u;                                        // count=1, user descriptor
  g0[1] = loff;                                      // lds_addr (bytes)
  g0[2] = ga_lo;                                     // global_addr[31:0]
  g0[3] = (ga_hi & 0x01FFFFFFu) | (2u << 30);        // global_addr[56:32] | type=2
  i32x8 g1;
  g1[0] = 0x00010000;                                // data_size = 2 bytes
  g1[1] = (int)((row_stride & 0xFFFFu) << 16);       // tensor_dim0[15:0]
  g1[2] = (int)((row_stride >> 16) | ((tile_h & 0xFFFFu) << 16)); // dim0 hi | dim1 lo
  g1[3] = (int)((tile_w & 0xFFFFu) << 16);           // dim1 hi(0) | tile_dim0
  g1[4] = (int)(tile_h & 0xFFFFu);                   // tile_dim1 | tile_dim2=0
  g1[5] = (int)row_stride;                           // tensor_dim0_stride[31:0]
  g1[6] = 0;                                         // stride0 hi | stride1 lo
  g1[7] = 0;                                         // stride1 hi
  i32x4 gz = {0, 0, 0, 0};
#if defined(__clang_major__) && (__clang_major__ >= 23)
  i32x8 gz8 = {0, 0, 0, 0, 0, 0, 0, 0};
  __builtin_amdgcn_tensor_load_to_lds(g0, g1, gz, gz, gz8, 0);
#else
  __builtin_amdgcn_tensor_load_to_lds(g0, g1, gz, gz, 0);
#endif
#else
  (void)lds_off; (void)tile_w; (void)tile_h; (void)row_stride;
  __builtin_prefetch(gaddr, 0, 1);
#endif
}

static __device__ __forceinline__ void tdm_wait_all() {
#if __has_builtin(__builtin_amdgcn_s_wait_tensorcnt)
  __builtin_amdgcn_s_wait_tensorcnt((short)0);
#endif
}

// ---------------------------------------------------------------------------
// Kernel 0: pre-pack W (fp32 [E][D]) -> f16 B-fragment layout.
// Wp index = ((which*32 + ec)*32 + lane)*128 + dt*16 + i
// element  = W[ec*32 + 16*(lane>>4) + i][dt*16 + (lane&15)]
// ---------------------------------------------------------------------------
__global__ __launch_bounds__(256) void wpack_kernel(
    const float* __restrict__ WQ,
    const float* __restrict__ WK,
    const float* __restrict__ WV,
    _Float16* __restrict__ Wp) {
  const int idx = blockIdx.x * 256 + threadIdx.x;   // 3*32*32*8*16 = 393216
  const int i     = idx & 15;
  const int dt    = (idx >> 4) & 7;
  const int lane  = (idx >> 7) & 31;
  const int ec    = (idx >> 12) & 31;
  const int which = idx >> 17;

  const float* W = (which == 0) ? WQ : (which == 1) ? WK : WV;
  const int e = ec * 32 + 16 * (lane >> 4) + i;
  const int d = dt * 16 + (lane & 15);
  Wp[idx] = (_Float16)W[(size_t)e * DD + d];
}

// ---------------------------------------------------------------------------
// Kernel 1: QKV projection. One wave computes a 16-row x 128-col tile of one
// of {Q,K,V}.  A = x tile (16x32 f16-converted), B = pre-packed W fragments
// (256 contiguous bytes per lane per chunk -> one load clause, 8 WMMAs).
// Q is pre-scaled by SCALE; V is stored transposed [B][D][S].
// ---------------------------------------------------------------------------
__global__ __launch_bounds__(128, 1) void qkv_proj_kernel(
    const float* __restrict__ x,
    const _Float16* __restrict__ Wp,
    _Float16* __restrict__ Qh,
    _Float16* __restrict__ Kh,
    _Float16* __restrict__ VhT) {
  const int lane  = threadIdx.x & 31;
  const int wv    = threadIdx.x >> 5;
  const int gwave = blockIdx.x * 4 + wv;     // 3*B*S/16 waves total
  const int which = gwave % 3;               // 0=Q 1=K 2=V (adjacent waves share x tile)
  const int tile  = gwave / 3;
  const int b     = tile / (SS / 16);
  const int s0    = (tile % (SS / 16)) * 16;
  const int lg    = lane >> 4;
  const int ln    = lane & 15;

  const float outScale = (which == 0) ? SCALE : 1.0f;
  const _Float16* wbase = Wp + (size_t)which * WP_PER_W + (size_t)lane * 128;

  v8f acc[8];
#pragma unroll
  for (int i = 0; i < 8; ++i)
#pragma unroll
    for (int j = 0; j < 8; ++j) acc[i][j] = 0.0f;

  const float* xrow = x + ((size_t)b * SS + (size_t)(s0 + ln)) * EE;

  for (int ec = 0; ec < ECHUNKS; ++ec) {
    // A fragment: rows m=ln, k(e): halfs 0..15 -> 16*lg..
    v16h a;
    {
      const float* p0 = xrow + ec * 32 + 16 * lg;
      const float* p1 = p0 + 8;
#pragma unroll
      for (int i = 0; i < 8; ++i) {
        a[i]     = (_Float16)p0[i];
        a[8 + i] = (_Float16)p1[i];
      }
    }
    // preload ALL 8 pre-packed B fragments (contiguous 256B), then 8 WMMAs
    const _Float16* wc = wbase + (size_t)ec * (32 * 128);
    v16h bm[8];
#pragma unroll
    for (int dt = 0; dt < 8; ++dt) {
      const h8 blo = *(const h8*)(wc + dt * 16);
      const h8 bhi = *(const h8*)(wc + dt * 16 + 8);
#pragma unroll
      for (int i = 0; i < 8; ++i) { bm[dt][i] = blo[i]; bm[dt][8 + i] = bhi[i]; }
    }
    sched_fence();  // pin the 16-load clause ahead of the WMMA burst
#pragma unroll
    for (int dt = 0; dt < 8; ++dt) acc[dt] = wmma_f16(a, bm[dt], acc[dt]);
  }

  if (which == 2) {
    _Float16* outp = VhT + (size_t)b * DD * SS;   // [D][S]
#pragma unroll
    for (int dt = 0; dt < 8; ++dt)
#pragma unroll
      for (int v = 0; v < 8; ++v)
        outp[(size_t)(dt * 16 + ln) * SS + (s0 + v + 8 * lg)] = (_Float16)acc[dt][v];
  } else {
    _Float16* outp = ((which == 0) ? Qh : Kh) + (size_t)b * SS * DD;  // [S][D]
#pragma unroll
    for (int dt = 0; dt < 8; ++dt)
#pragma unroll
      for (int v = 0; v < 8; ++v)
        outp[(size_t)(s0 + v + 8 * lg) * DD + dt * 16 + ln] =
            (_Float16)(acc[dt][v] * outScale);
  }
}

// ---------------------------------------------------------------------------
// Kernel 2: flash attention, one wave per 16-query tile, 32-key blocks.
// ---------------------------------------------------------------------------
__global__ __launch_bounds__(128, 1) void flash_attn_kernel(
    const _Float16* __restrict__ Qh,
    const _Float16* __restrict__ Kh,
    const _Float16* __restrict__ VhT,
    float* __restrict__ out) {
  __shared__ _Float16 ldsP[4][16 * 32];    // per-wave 1KB P^T staging slab
  __shared__ _Float16 tdmK[32 * DD];       // TDM prefetch target, next K block
  __shared__ _Float16 tdmV[DD * 32];       // TDM prefetch target, next V^T block

  const int lane = threadIdx.x & 31;
  const int wv   = threadIdx.x >> 5;
  const int gw   = blockIdx.x * 4 + wv;    // B*S/16 waves
  const int b    = gw / (SS / 16);
  const int q0   = (gw % (SS / 16)) * 16;
  const int lg   = lane >> 4;
  const int ln   = lane & 15;

  const _Float16* Qb = Qh + (size_t)b * SS * DD;
  const _Float16* Kb = Kh + (size_t)b * SS * DD;
  const _Float16* Vb = VhT + (size_t)b * DD * SS;

  const u32 koff = (u32)(size_t)&tdmK[0];
  const u32 voff = (u32)(size_t)&tdmV[0];

  // Q^T B-fragments, one per 32-wide d-chunk: b[i] = Q[q0+ln][dc*32 + 16*lg + i]
  v16h qf[4];
#pragma unroll
  for (int dc = 0; dc < 4; ++dc) {
    const _Float16* p = Qb + (size_t)(q0 + ln) * DD + dc * 32 + 16 * lg;
    const h8 lo = *(const h8*)p;
    const h8 hi = *(const h8*)(p + 8);
#pragma unroll
    for (int i = 0; i < 8; ++i) { qf[dc][i] = lo[i]; qf[dc][8 + i] = hi[i]; }
  }

  v8f oacc[8];
#pragma unroll
  for (int i = 0; i < 8; ++i)
#pragma unroll
    for (int j = 0; j < 8; ++j) oacc[i][j] = 0.0f;

  float m_i = -1e30f;   // running row max for query q0+ln
  float l_i = 0.0f;     // running row sum

  const int kend = q0 + 16;  // keys needed: [0, q0+15]
  for (int kb0 = 0; kb0 < kend; kb0 += 32) {
    // ---- TDM: pull next K (32x128, stride D) and V^T (128x32, stride S)
    //      blocks toward LDS/L2 while this block computes ----
    if (kb0 + 32 < kend) {
      tdm_load_2d(Kb + (size_t)(kb0 + 32) * DD, koff, /*w*/DD, /*h*/32, /*stride*/DD);
      tdm_load_2d(Vb + (kb0 + 32),              voff, /*w*/32, /*h*/DD, /*stride*/SS);
    }

    // ---- preload K fragments (8) and V fragments (8); V loads overlap
    //      GEMM1 + softmax ----
    v16h ka[8];  // [mt*4 + dc]
#pragma unroll
    for (int mt = 0; mt < 2; ++mt)
#pragma unroll
      for (int dc = 0; dc < 4; ++dc) {
        const _Float16* p0 = Kb + (size_t)(kb0 + mt * 16 + ln) * DD + dc * 32 + 16 * lg;
        const h8 alo = *(const h8*)p0;
        const h8 ahi = *(const h8*)(p0 + 8);
        v16h* kf = &ka[mt * 4 + dc];
#pragma unroll
        for (int i = 0; i < 8; ++i) { (*kf)[i] = alo[i]; (*kf)[8 + i] = ahi[i]; }
      }
    v16h vbf[8];
#pragma unroll
    for (int dt = 0; dt < 8; ++dt) {
      const _Float16* vp = Vb + (size_t)(dt * 16 + ln) * SS + kb0 + 16 * lg;
      const h8 vlo = *(const h8*)vp;
      const h8 vhi = *(const h8*)(vp + 8);
#pragma unroll
      for (int i = 0; i < 8; ++i) { vbf[dt][i] = vlo[i]; vbf[dt][8 + i] = vhi[i]; }
    }
    sched_fence();  // keep all 32 fragment loads issued before compute

    // ---- GEMM1: S^T tile = K(32x128) @ Q^T(128x16), two 16-key M-tiles ----
    v8f sc[2];
#pragma unroll
    for (int mt = 0; mt < 2; ++mt) {
#pragma unroll
      for (int j = 0; j < 8; ++j) sc[mt][j] = 0.0f;
#pragma unroll
      for (int dc = 0; dc < 4; ++dc) sc[mt] = wmma_f16(ka[mt * 4 + dc], qf[dc], sc[mt]);
    }

    // ---- causal mask (scale already folded into Q) ----
    if (kb0 + 31 > q0) {
#pragma unroll
      for (int mt = 0; mt < 2; ++mt)
#pragma unroll
        for (int v = 0; v < 8; ++v) {
          const int key = kb0 + mt * 16 + v + 8 * lg;
          if (key > q0 + ln) sc[mt][v] = -1e30f;
        }
    }

    // ---- online softmax: all values in a lane belong to query q0+ln ----
    float mloc = m_i;
#pragma unroll
    for (int mt = 0; mt < 2; ++mt)
#pragma unroll
      for (int v = 0; v < 8; ++v) mloc = fmaxf(mloc, sc[mt][v]);
    mloc = fmaxf(mloc, __shfl_xor(mloc, 16));

    const float alpha = __expf(m_i - mloc);
    float rsum = 0.0f;
#pragma unroll
    for (int mt = 0; mt < 2; ++mt)
#pragma unroll
      for (int v = 0; v < 8; ++v) {
        const float p = __expf(sc[mt][v] - mloc);
        sc[mt][v] = p;
        rsum += p;
      }
    rsum += __shfl_xor(rsum, 16);
    l_i = l_i * alpha + rsum;
    m_i = mloc;

    // rescale accumulator rows: oacc element (lane,v) is row q = v+8*lg
    float av[8];
#pragma unroll
    for (int v = 0; v < 8; ++v) av[v] = __shfl(alpha, v + 8 * lg);
#pragma unroll
    for (int dt = 0; dt < 8; ++dt)
#pragma unroll
      for (int v = 0; v < 8; ++v) oacc[dt][v] *= av[v];

    // ---- reformat P^T (C layout) -> P A-fragment via LDS ----
    _Float16* slab = ldsP[wv];
#pragma unroll
    for (int mt = 0; mt < 2; ++mt)
#pragma unroll
      for (int v = 0; v < 8; ++v)
        slab[ln * 32 + mt * 16 + v + 8 * lg] = (_Float16)sc[mt][v];
    asm volatile("s_wait_dscnt 0x0" ::: "memory");

    v16h pa;  // A fragment of P (16 q x 32 k)
    {
      const _Float16* pr = slab + ln * 32;
      const h8 plo = *(const h8*)(pr + 16 * lg);
      const h8 phi = *(const h8*)(pr + 8 + 16 * lg);
#pragma unroll
      for (int i = 0; i < 8; ++i) { pa[i] = plo[i]; pa[8 + i] = phi[i]; }
    }

    // ---- GEMM2: O += P(16x32) @ V(32x128), fragments already resident ----
#pragma unroll
    for (int dt = 0; dt < 8; ++dt) oacc[dt] = wmma_f16(pa, vbf[dt], oacc[dt]);
  }

  tdm_wait_all();  // drain TENSORcnt before wave exit

  // ---- epilogue: multiply by reciprocal row sums, store fp32 ----
  const float rli = __frcp_rn(l_i);
  float lv[8];
#pragma unroll
  for (int v = 0; v < 8; ++v) lv[v] = __shfl(rli, v + 8 * lg);

  float* outp = out + ((size_t)b * SS + q0) * DD;
#pragma unroll
  for (int dt = 0; dt < 8; ++dt)
#pragma unroll
    for (int v = 0; v < 8; ++v)
      outp[(size_t)(v + 8 * lg) * DD + dt * 16 + ln] = oacc[dt][v] * lv[v];
}

// ---------------------------------------------------------------------------
extern "C" void kernel_launch(void* const* d_in, const int* in_sizes, int n_in,
                              void* d_out, int out_size, void* d_ws, size_t ws_size,
                              hipStream_t stream) {
  const float* x  = (const float*)d_in[0];
  const float* WQ = (const float*)d_in[1];
  const float* WK = (const float*)d_in[2];
  const float* WV = (const float*)d_in[3];
  float* out = (float*)d_out;

  const size_t tensElems = (size_t)BB * SS * DD;  // 2M f16 each
  _Float16* Qh  = (_Float16*)d_ws;
  _Float16* Kh  = Qh + tensElems;
  _Float16* VhT = Kh + tensElems;
  _Float16* Wp  = VhT + tensElems;  // +768 KB packed weights (12.75 MB total)

  // Phase 0: pack weights into WMMA B-fragment layout (3*131072 halfs)
  wpack_kernel<<<dim3(3 * WP_PER_W / 256), dim3(256), 0, stream>>>(WQ, WK, WV, Wp);

  // Phase 1: 3 * B * S/16 = 3072 waves, 4 waves/block
  qkv_proj_kernel<<<dim3(3 * BB * (SS / 16) / 4), dim3(128), 0, stream>>>(
      x, Wp, Qh, Kh, VhT);

  // Phase 2: B * S/16 = 1024 waves, 4 waves/block
  flash_attn_kernel<<<dim3(BB * (SS / 16) / 4), dim3(128), 0, stream>>>(
      Qh, Kh, VhT, out);
}
